// Attention_9431748182617
// MI455X (gfx1250) — matrix operand order repaired and checked by reading, not compiled, not run on hardware
//
#include <hip/hip_runtime.h>
#include <hip/hip_bf16.h>

// Problem constants (from the reference)
#define BQ   4
#define SQ   2048
#define DQ   1024
#define LQ   1024
#define HQ   16
#define DHQ  64
#define SCALE_Q 0.03125f   // 1/sqrt(1024)

typedef __attribute__((ext_vector_type(16))) __bf16 v16bf;
typedef __attribute__((ext_vector_type(8)))  float  v8f;
typedef __attribute__((ext_vector_type(4)))  int    v4i;

union FragBF { v16bf v; uint4 q[2]; };

static __device__ __forceinline__ v8f wmma_bf16(const FragBF& a, const FragBF& b, v8f c) {
  // (neg_a, A, neg_b, B, c_mod, C, reuse_a, reuse_b)
  return __builtin_amdgcn_wmma_f32_16x16x32_bf16(false, a.v, false, b.v, (short)0, c, false, false);
}

static __device__ __forceinline__ v8f zero8() {
  v8f z = {0.f,0.f,0.f,0.f,0.f,0.f,0.f,0.f};
  return z;
}

static __device__ __forceinline__ void store_out(float* p, float v)  { *p = v; }
static __device__ __forceinline__ void store_out(__bf16* p, float v) { *p = (__bf16)v; }

// ---------------------------------------------------------------------------
// gfx1250 async global->LDS copy (ASYNCcnt-tracked), with safe fallback.
// Builtin signature (from hipcc diagnostic): (v4i AS1* src, v4i AS3* dst,
// imm int offset, imm int cpol).
// ---------------------------------------------------------------------------
#if defined(__has_builtin)
#if __has_builtin(__builtin_amdgcn_global_load_async_to_lds_b128) && \
    __has_builtin(__builtin_amdgcn_s_wait_asynccnt)
#define HAS_ASYNC_LDS 1
#endif
#endif
#ifndef HAS_ASYNC_LDS
#define HAS_ASYNC_LDS 0
#endif

static __device__ __forceinline__ void async_ld_b128(void* lds_dst, const void* gsrc) {
#if HAS_ASYNC_LDS
  __builtin_amdgcn_global_load_async_to_lds_b128(
      (__attribute__((address_space(1))) v4i*)(gsrc),
      (__attribute__((address_space(3))) v4i*)(lds_dst), 0, 0);
#else
  *(uint4*)lds_dst = *(const uint4*)gsrc;
#endif
}
static __device__ __forceinline__ void async_wait0() {
#if HAS_ASYNC_LDS
  __builtin_amdgcn_s_wait_asynccnt(0);
#endif
}

// ---------------------------------------------------------------------------
// f32 -> bf16 conversion
// ---------------------------------------------------------------------------
__global__ __launch_bounds__(256) void k_f32_to_bf16(const float* __restrict__ in,
                                                     __bf16* __restrict__ out, int n) {
  int i = blockIdx.x * blockDim.x + threadIdx.x;
  if (i < n) out[i] = (__bf16)in[i];
}

// ---------------------------------------------------------------------------
// NT GEMM: C[M,N] = A[M,K] * W[N,K]^T  (A,W bf16 row-major, f32 accumulate)
// Block 256 thr = 8 waves. Block tile 128x128, wave tile 64x32, BK = 32.
// Double-buffered LDS; A tile staged by async DMA, W tile transposed on store.
// ---------------------------------------------------------------------------
template <typename OutT>
__global__ __launch_bounds__(256) void k_gemm_nt(OutT* __restrict__ C,
                                                 const __bf16* __restrict__ A,
                                                 const __bf16* __restrict__ W,
                                                 int M, int N, int K) {
  __shared__ __align__(16) __bf16 As[2][128][32];
  __shared__ __align__(16) __bf16 Wt[2][32][128];   // Wt[k][n]

  const int t    = threadIdx.x;
  const int lane = t & 31;
  const int w    = t >> 5;
  const int r    = lane & 15;
  const int h    = lane >> 4;
  const int wm   = w >> 2;       // 0..1
  const int wn   = w & 3;        // 0..3
  const int m0   = blockIdx.y * 128;
  const int n0   = blockIdx.x * 128;

  v8f acc[4][2];
#pragma unroll
  for (int i = 0; i < 4; ++i)
#pragma unroll
    for (int j = 0; j < 2; ++j) acc[i][j] = zero8();

  const int lrow = t >> 1, lhf = t & 1;
  const __bf16* aRow = A + (size_t)(m0 + lrow) * K + lhf * 16;
  const __bf16* wRow = W + (size_t)(n0 + lrow) * K + lhf * 16;

  // Prologue: fill buffer 0 for k0 = 0
  {
    async_ld_b128(&As[0][lrow][lhf * 16],     aRow);
    async_ld_b128(&As[0][lrow][lhf * 16 + 8], aRow + 8);
    union { uint4 u[2]; __bf16 e[16]; } wt;
    wt.u[0] = *(const uint4*)(wRow);
    wt.u[1] = *(const uint4*)(wRow + 8);
#pragma unroll
    for (int i2 = 0; i2 < 16; ++i2) Wt[0][lhf * 16 + i2][lrow] = wt.e[i2];
    async_wait0();
    __syncthreads();
  }

  const int NS = K / 32;
  for (int s = 0; s < NS; ++s) {
    const int buf = s & 1;
    const bool more = (s + 1 < NS);
    union { uint4 u[2]; __bf16 e[16]; } wt;
    if (more) {
      const int kn = (s + 1) * 32;
      // Prefetch next A tile straight into the other LDS buffer (async DMA).
      async_ld_b128(&As[buf ^ 1][lrow][lhf * 16],     aRow + kn);
      async_ld_b128(&As[buf ^ 1][lrow][lhf * 16 + 8], aRow + kn + 8);
      // Prefetch next W tile into registers.
      wt.u[0] = *(const uint4*)(wRow + kn);
      wt.u[1] = *(const uint4*)(wRow + kn + 8);
    }

    // Compute on current buffer
    FragBF a[4], bf[2];
#pragma unroll
    for (int i = 0; i < 4; ++i) {
      const __bf16* p = &As[buf][wm * 64 + i * 16 + r][h * 8];
      a[i].q[0] = *(const uint4*)p;
      a[i].q[1] = *(const uint4*)(p + 16);
    }
#pragma unroll
    for (int j = 0; j < 2; ++j) {
#pragma unroll
      for (int s2 = 0; s2 < 16; ++s2)
        bf[j].v[s2] = Wt[buf][(s2 & 7) + 8 * h + 16 * (s2 >> 3)][wn * 32 + j * 16 + r];
    }
#pragma unroll
    for (int i = 0; i < 4; ++i)
#pragma unroll
      for (int j = 0; j < 2; ++j) acc[i][j] = wmma_bf16(a[i], bf[j], acc[i][j]);

    if (more) {
#pragma unroll
      for (int i2 = 0; i2 < 16; ++i2) Wt[buf ^ 1][lhf * 16 + i2][lrow] = wt.e[i2];
      async_wait0();
    }
    __syncthreads();
  }

#pragma unroll
  for (int i = 0; i < 4; ++i)
#pragma unroll
    for (int j = 0; j < 2; ++j) {
      const int n = n0 + wn * 32 + j * 16 + r;
#pragma unroll
      for (int jr = 0; jr < 8; ++jr) {
        const int m = m0 + wm * 64 + i * 16 + 8 * h + jr;
        store_out(C + (size_t)m * N + n, acc[i][j][jr]);
      }
    }
}

// ---------------------------------------------------------------------------
// Pass 1: per-(head, key) softmax stats over the QUERY axis.
// Each lane keeps PRIVATE partial (max, sum) over its q-subset (q % 16 == r);
// one cross-lane merge at the very end. Stats stored as (m_k, 1/Z_k).
// ---------------------------------------------------------------------------
__global__ __launch_bounds__(256) void k_softmax_stats(const __bf16* __restrict__ Qb,
                                                       const __bf16* __restrict__ Kb,
                                                       float2* __restrict__ stats) {
  __shared__ __align__(16) __bf16 Qt[64][128];   // Qt[d][q] for a 128-query chunk

  const int t    = threadIdx.x;
  const int lane = t & 31;
  const int w    = t >> 5;
  const int r    = lane & 15;
  const int h    = lane >> 4;
  const int kblk = blockIdx.x;           // 0..15 (128 k-rows each)
  const int bh   = blockIdx.y;           // 0..63
  const int b    = bh >> 4, hh = bh & 15;
  const size_t headOff = (size_t)b * SQ * LQ + (size_t)hh * DHQ;

  // A-frags: 16 K-rows x 32 d, two d-steps (d = 0..31, 32..63). k-invariant.
  const int krow = kblk * 128 + w * 16 + r;
  FragBF aK[2];
#pragma unroll
  for (int s = 0; s < 2; ++s) {
    const __bf16* kp = Kb + headOff + (size_t)krow * LQ + s * 32 + h * 8;
    aK[s].q[0] = *(const uint4*)kp;
    aK[s].q[1] = *(const uint4*)(kp + 16);
  }

  float m[8], z[8];
#pragma unroll
  for (int j = 0; j < 8; ++j) { m[j] = -1e30f; z[j] = 0.f; }

  const int qrow = t >> 1, qh = t & 1;
  for (int qc = 0; qc < SQ; qc += 128) {
    // Stage Q chunk transposed: Qt[d][q]
    union { uint4 u[4]; __bf16 e[32]; } qt;
    const __bf16* qp = Qb + headOff + (size_t)(qc + qrow) * LQ + qh * 32;
    qt.u[0] = *(const uint4*)(qp);
    qt.u[1] = *(const uint4*)(qp + 8);
    qt.u[2] = *(const uint4*)(qp + 16);
    qt.u[3] = *(const uint4*)(qp + 24);
#pragma unroll
    for (int i2 = 0; i2 < 32; ++i2) Qt[qh * 32 + i2][qrow] = qt.e[i2];
    __syncthreads();

    for (int qt16 = 0; qt16 < 8; ++qt16) {
      FragBF b0, b1;
      {
#pragma unroll
        for (int s2 = 0; s2 < 16; ++s2) {
          const int kk = (s2 & 7) + 8 * h + 16 * (s2 >> 3);
          b0.v[s2] = Qt[kk][qt16 * 16 + r];
          b1.v[s2] = Qt[32 + kk][qt16 * 16 + r];
        }
      }
      v8f sc = zero8();
      sc = wmma_bf16(aK[0], b0, sc);
      sc = wmma_bf16(aK[1], b1, sc);
      // Lane-private online softmax update: ONE exp per element, no shuffles.
#pragma unroll
      for (int j = 0; j < 8; ++j) {
        const float v  = sc[j] * SCALE_Q;
        const float e  = __expf(-fabsf(v - m[j]));
        const bool  gt = (v > m[j]);
        z[j] = gt ? (z[j] * e + 1.f) : (z[j] + e);
        m[j] = gt ? v : m[j];
      }
    }
    __syncthreads();
  }

  // Merge lane-partial (m, z) across the 16-lane N-group (once).
#pragma unroll
  for (int j = 0; j < 8; ++j) {
#pragma unroll
    for (int off = 1; off < 16; off <<= 1) {
      const float mo = __shfl_xor(m[j], off, 32);
      const float zo = __shfl_xor(z[j], off, 32);
      const float mn = fmaxf(m[j], mo);
      z[j] = z[j] * __expf(m[j] - mn) + zo * __expf(mo - mn);
      m[j] = mn;
    }
  }

  if (r == 0) {
#pragma unroll
    for (int j = 0; j < 8; ++j) {
      const int kk = kblk * 128 + w * 16 + j + 8 * h;
      stats[(size_t)bh * SQ + kk] = make_float2(m[j], 1.f / z[j]);  // (max, 1/Z)
    }
  }
}

// ---------------------------------------------------------------------------
// Pass 2: out[q,d] = sum_k exp(s[k,q]-m_k) * (1/Z_k) * V[k,d].
// Scores recomputed by WMMA; weight tile transposed via ONE packed b128 LDS
// store per lane per kt; stats read as aligned float4 pairs.
// ---------------------------------------------------------------------------
__global__ __launch_bounds__(256) void k_attn_out(const __bf16* __restrict__ Qb,
                                                  const __bf16* __restrict__ Kb,
                                                  const __bf16* __restrict__ Vb,
                                                  const float2* __restrict__ stats,
                                                  __bf16* __restrict__ Z) {
  __shared__ __align__(16) __bf16 Qt[64][128];
  __shared__ __align__(16) float2 sLDS[SQ];
  __shared__ __align__(16) __bf16 wlds[8][16][32];   // per-wave transpose buffer

  const int t    = threadIdx.x;
  const int lane = t & 31;
  const int w    = t >> 5;
  const int r    = lane & 15;
  const int h    = lane >> 4;
  const int qblk = blockIdx.x;           // 0..15
  const int bh   = blockIdx.y;           // 0..63
  const int b    = bh >> 4, hh = bh & 15;
  const size_t headOff = (size_t)b * SQ * LQ + (size_t)hh * DHQ;

  // Stage softmax stats (m, 1/Z) for this head
#pragma unroll
  for (int i = 0; i < 8; ++i) {
    const int idx = t + i * 256;
    sLDS[idx] = stats[(size_t)bh * SQ + idx];
  }
  // Stage Q block transposed: Qt[d][q]
  {
    const int qrow = t >> 1, qh = t & 1;
    union { uint4 u[4]; __bf16 e[32]; } qt;
    const __bf16* qp = Qb + headOff + (size_t)(qblk * 128 + qrow) * LQ + qh * 32;
    qt.u[0] = *(const uint4*)(qp);
    qt.u[1] = *(const uint4*)(qp + 8);
    qt.u[2] = *(const uint4*)(qp + 16);
    qt.u[3] = *(const uint4*)(qp + 24);
#pragma unroll
    for (int i2 = 0; i2 < 32; ++i2) Qt[qh * 32 + i2][qrow] = qt.e[i2];
  }
  __syncthreads();

  // k-invariant score B-frags for this wave's 16 queries
  FragBF bQ0, bQ1;
  {
#pragma unroll
    for (int s2 = 0; s2 < 16; ++s2) {
      const int kk = (s2 & 7) + 8 * h + 16 * (s2 >> 3);
      bQ0.v[s2] = Qt[kk][w * 16 + r];
      bQ1.v[s2] = Qt[32 + kk][w * 16 + r];
    }
  }

  v8f accO[4];
#pragma unroll
  for (int n = 0; n < 4; ++n) accO[n] = zero8();

  for (int k0 = 0; k0 < SQ; k0 += 32) {
#pragma unroll
    for (int kt = 0; kt < 2; ++kt) {
      const int kr = k0 + kt * 16 + r;
      FragBF a0, a1;
      const __bf16* kp0 = Kb + headOff + (size_t)kr * LQ + h * 8;
      a0.q[0] = *(const uint4*)kp0;        a0.q[1] = *(const uint4*)(kp0 + 16);
      const __bf16* kp1 = kp0 + 32;
      a1.q[0] = *(const uint4*)kp1;        a1.q[1] = *(const uint4*)(kp1 + 16);
      v8f sc = zero8();
      sc = wmma_bf16(a0, bQ0, sc);
      sc = wmma_bf16(a1, bQ1, sc);

      // Rows handled by this lane group: kk0 .. kk0+7 (contiguous)
      const int kk0 = k0 + kt * 16 + 8 * h;
      const float4* sp = (const float4*)&sLDS[kk0];   // (m,1/Z) pairs, 16B aligned
      union { uint4 u; __bf16 e[8]; } pk;
#pragma unroll
      for (int jj = 0; jj < 4; ++jj) {
        const float4 mz2 = sp[jj];                    // m_j, r_j, m_{j+1}, r_{j+1}
        const float w0 = __expf(sc[2 * jj]     * SCALE_Q - mz2.x) * mz2.y;
        const float w1 = __expf(sc[2 * jj + 1] * SCALE_Q - mz2.z) * mz2.w;
        pk.e[2 * jj]     = (__bf16)w0;
        pk.e[2 * jj + 1] = (__bf16)w1;
      }
      // One packed 16B store transposes this lane's 8 weights (k contiguous).
      *(uint4*)&wlds[w][r][kt * 16 + 8 * h] = pk.u;
    }
    // Read attention weights back in WMMA-A layout: A[q][k] 16x32
    FragBF aW;
    {
      const __bf16* p = &wlds[w][r][h * 8];
      aW.q[0] = *(const uint4*)p;
      aW.q[1] = *(const uint4*)(p + 16);
    }
#pragma unroll
    for (int n = 0; n < 4; ++n) {
      FragBF bV;
#pragma unroll
      for (int s2 = 0; s2 < 16; ++s2)
        bV.v[s2] = Vb[headOff + (size_t)(k0 + (s2 & 7) + 8 * h + 16 * (s2 >> 3)) * LQ + n * 16 + r];
      accO[n] = wmma_bf16(aW, bV, accO[n]);
    }
  }

#pragma unroll
  for (int n = 0; n < 4; ++n)
#pragma unroll
    for (int j = 0; j < 8; ++j) {
      const int q = qblk * 128 + w * 16 + j + 8 * h;
      Z[headOff + (size_t)q * LQ + n * 16 + r] = (__bf16)accO[n][j];
    }
}

// ---------------------------------------------------------------------------
// Launch
// ---------------------------------------------------------------------------
extern "C" void kernel_launch(void* const* d_in, const int* in_sizes, int n_in,
                              void* d_out, int out_size, void* d_ws, size_t ws_size,
                              hipStream_t stream) {
  (void)in_sizes; (void)n_in; (void)out_size; (void)ws_size;
  const float* x  = (const float*)d_in[0];
  const float* Wq = (const float*)d_in[1];
  const float* Wk = (const float*)d_in[2];
  const float* Wv = (const float*)d_in[3];
  const float* Wo = (const float*)d_in[4];
  float* out = (float*)d_out;

  const int MROWS = BQ * SQ;              // 8192
  const size_t nX = (size_t)MROWS * DQ;   // 8,388,608
  const size_t nW = (size_t)LQ * DQ;      // 1,048,576

  char* p = (char*)d_ws;
  __bf16* Xb  = (__bf16*)p; p += nX * 2;
  __bf16* Wqb = (__bf16*)p; p += nW * 2;
  __bf16* Wkb = (__bf16*)p; p += nW * 2;
  __bf16* Wvb = (__bf16*)p; p += nW * 2;
  __bf16* Wob = (__bf16*)p; p += nW * 2;
  __bf16* Qb  = (__bf16*)p; p += nX * 2;
  __bf16* Kb  = (__bf16*)p; p += nX * 2;
  __bf16* Vb  = (__bf16*)p; p += nX * 2;
  __bf16* Zb  = (__bf16*)p; p += nX * 2;
  float2* st  = (float2*)p; p += (size_t)BQ * HQ * SQ * sizeof(float2);

  // 1) f32 -> bf16 conversions
  k_f32_to_bf16<<<(nX + 255) / 256, 256, 0, stream>>>(x,  Xb,  (int)nX);
  k_f32_to_bf16<<<(nW + 255) / 256, 256, 0, stream>>>(Wq, Wqb, (int)nW);
  k_f32_to_bf16<<<(nW + 255) / 256, 256, 0, stream>>>(Wk, Wkb, (int)nW);
  k_f32_to_bf16<<<(nW + 255) / 256, 256, 0, stream>>>(Wv, Wvb, (int)nW);
  k_f32_to_bf16<<<(nW + 255) / 256, 256, 0, stream>>>(Wo, Wob, (int)nW);

  // 2) QKV projections: (8192x1024) = X (8192x1024) @ W^T (1024x1024)
  dim3 gGemm(LQ / 128, MROWS / 128);   // (8, 64)
  k_gemm_nt<__bf16><<<gGemm, 256, 0, stream>>>(Qb, Xb, Wqb, MROWS, LQ, DQ);
  k_gemm_nt<__bf16><<<gGemm, 256, 0, stream>>>(Kb, Xb, Wkb, MROWS, LQ, DQ);
  k_gemm_nt<__bf16><<<gGemm, 256, 0, stream>>>(Vb, Xb, Wvb, MROWS, LQ, DQ);

  // 3) Softmax stats over queries (per head, per key)
  dim3 gAttn(SQ / 128, BQ * HQ);       // (16, 64)
  k_softmax_stats<<<gAttn, 256, 0, stream>>>(Qb, Kb, st);

  // 4) Attention output: Z[q,d] = sum_k P[k,q] V[k,d]
  k_attn_out<<<gAttn, 256, 0, stream>>>(Qb, Kb, Vb, st, Zb);

  // 5) Output projection (f32 result): out = Z (8192x1024) @ Wo^T
  dim3 gOut(DQ / 128, MROWS / 128);
  k_gemm_nt<float><<<gOut, 256, 0, stream>>>(out, Zb, Wob, MROWS, DQ, LQ);
}
